// NeighborhoodAttention_30777735643437
// MI455X (gfx1250) — matrix operand order repaired and checked by reading, not compiled, not run on hardware
//
#include <hip/hip_runtime.h>
#include <hip/hip_bf16.h>

// ---------------- problem constants ----------------
#define BB      2
#define CC      192
#define HH      64
#define WW      64
#define HEADS   6
#define HD      32
#define KS      7
#define HWSZ    (HH * WW)              // 4096
#define NN      (BB * HWSZ)            // 8192 pixels total
#define C3      (3 * CC)               // 576
#define NT_ALL  (NN / 16)              // 512 n-tiles

typedef __attribute__((ext_vector_type(16))) _Float16 v16h;
typedef __attribute__((ext_vector_type(8)))  float    v8f;

union CFrag { v8f v; float f[8]; };

// ===========================================================================
// Fragment-ready packed layouts (so the GEMM loads are contiguous b128):
//   A-pack  (weights [M][K]):  off = ((kt*Mt + mt)*32 + lane)*16 + e
//       m  = mt*16 + (lane&15)
//       kb = (lane>>4)*16 + e
//       k  = kt*32 + kb                       (contiguous k order, same as B-pack)
//   B-pack  (activations [K][N]): off = ((kt*Nt + nt)*32 + lane)*16 + e
//       n  = nt*16 + (lane&15)
//       k  = kt*32 + (lane>>4)*16 + e        (contiguous k order, same as A-pack)
// ===========================================================================

// Pack fp32 row-major weights [M][K] into WMMA A-fragment layout (f16)
__global__ void na_pack_wA(const float* __restrict__ Wsrc, _Float16* __restrict__ Ap,
                           int M, int K) {
    int idx = blockIdx.x * blockDim.x + threadIdx.x;
    if (idx >= M * K) return;
    const int e    = idx & 15;
    const int l    = (idx >> 4) & 31;
    const int tile = idx >> 9;
    const int Mt   = M >> 4;
    const int mt   = tile % Mt;
    const int kt   = tile / Mt;
    const int m    = mt * 16 + (l & 15);
    const int v    = e >> 1;
    const int kb   = ((l >> 4) << 4) + e;
    const int k    = kt * 32 + kb;
    Ap[idx] = (_Float16)Wsrc[(size_t)m * K + k];
}

// Pack x (B,C,H,W) fp32 directly into WMMA B-fragment layout (f16), K=C, N=B*HW
__global__ void na_pack_xB(const float* __restrict__ x, _Float16* __restrict__ Bp) {
    int idx = blockIdx.x * blockDim.x + threadIdx.x;
    if (idx >= CC * NN) return;
    const int e    = idx & 15;
    const int l    = (idx >> 4) & 31;
    const int tile = idx >> 9;
    const int nt   = tile % NT_ALL;
    const int kt   = tile / NT_ALL;
    const int n    = nt * 16 + (l & 15);
    const int k    = kt * 32 + ((l >> 4) << 4) + e;
    const int b    = n >> 12;                 // / HWSZ
    const int hw   = n & (HWSZ - 1);
    Bp[idx] = (_Float16)x[((size_t)b * CC + k) * HWSZ + hw];
}

// ---------------------------------------------------------------------------
// WMMA GEMM on pre-packed fragments. One wave owns a 16x64 macro-tile:
// A fragment loaded once per K-step (2x b128), reused for 4 wmma ops.
// mode 0: D[M][N] scratch layout;  mode 1: final NCHW output layout.
// ---------------------------------------------------------------------------
__global__ void na_gemm_wmma(const _Float16* __restrict__ Ap,
                             const _Float16* __restrict__ Bp,
                             const float*    __restrict__ bias,
                             float*          __restrict__ D,
                             int M, int N, int K, int mode)
{
    const int tid  = blockIdx.x * blockDim.x + threadIdx.x;
    const int wave = tid >> 5;
    const int lane = tid & 31;
    const int Mt   = M >> 4;
    const int Nt   = N >> 4;
    const int Ng   = Nt >> 2;                 // groups of 4 n-tiles
    if (wave >= Mt * Ng) return;
    const int mt = wave / Ng;
    const int ng = wave % Ng;
    const int Kt = K >> 5;

    CFrag c[4];
#pragma unroll
    for (int t = 0; t < 4; ++t)
#pragma unroll
        for (int r = 0; r < 8; ++r) c[t].f[r] = 0.0f;

    for (int kt = 0; kt < Kt; ++kt) {
        const _Float16* ap = Ap + (((size_t)kt * Mt + mt) * 32 + lane) * 16;
        const v16h a = *(const v16h*)ap;
        const _Float16* bp0 = Bp + (((size_t)kt * Nt + (ng << 2)) * 32 + lane) * 16;
        if (kt + 1 < Kt) {                    // -> global_prefetch_b8
            __builtin_prefetch(ap + (size_t)Mt * 512, 0, 3);
            __builtin_prefetch(bp0 + (size_t)Nt * 512, 0, 3);
        }
#pragma unroll
        for (int t = 0; t < 4; ++t) {
            const v16h b = *(const v16h*)(bp0 + (size_t)t * 512);
            c[t].v = __builtin_amdgcn_wmma_f32_16x16x32_f16(
                         false, a, false, b, (short)0, c[t].v, false, false);
        }
    }

    // C/D layout: VGPR r -> M = mt*16 + r + (lane>>4)*8, N = nt*16 + (lane&15)
    const int hi = lane >> 4;
    const int lo = lane & 15;
#pragma unroll
    for (int t = 0; t < 4; ++t) {
        const int n = ((ng << 2) + t) * 16 + lo;
#pragma unroll
        for (int r = 0; r < 8; ++r) {
            const int m   = mt * 16 + r + (hi << 3);
            const float v = c[t].f[r] + bias[m];
            if (mode == 0) {
                D[(size_t)m * N + n] = v;
            } else {
                const int b  = n / HWSZ;
                const int hw = n % HWSZ;
                D[((size_t)b * M + m) * HWSZ + hw] = v;
            }
        }
    }
}

// ---------------------------------------------------------------------------
// Depthwise 3x3 SAME conv + bias on QKV1[C3][N] -> QKV2[C3][N]
// ---------------------------------------------------------------------------
__global__ void na_dwconv3x3(const float* __restrict__ Q1,
                             const float* __restrict__ Wd,
                             const float* __restrict__ bd,
                             float*       __restrict__ Q2)
{
    int idx = blockIdx.x * blockDim.x + threadIdx.x;
    if (idx >= C3 * NN) return;
    const int n  = idx % NN;
    const int co = idx / NN;
    const int b  = n >> 12;
    const int hw = n & (HWSZ - 1);
    const int i  = hw >> 6;
    const int j  = hw & 63;

    float acc = bd[co];
    const float* base = Q1 + (size_t)co * NN + (size_t)b * HWSZ;
#pragma unroll
    for (int di = 0; di < 3; ++di) {
#pragma unroll
        for (int dj = 0; dj < 3; ++dj) {
            const int ii = i + di - 1;
            const int jj = j + dj - 1;
            if (ii >= 0 && ii < HH && jj >= 0 && jj < WW)
                acc += base[ii * WW + jj] * Wd[co * 9 + di * 3 + dj];
        }
    }
    Q2[(size_t)co * NN + n] = acc;
}

// ---------------------------------------------------------------------------
// In-place L2 normalization of q and k rows over HD channels.
// ---------------------------------------------------------------------------
__global__ void na_l2norm(float* __restrict__ Q2)
{
    int idx = blockIdx.x * blockDim.x + threadIdx.x;
    if (idx >= 2 * HEADS * NN) return;
    const int n    = idx % NN;
    const int base = (idx / NN) * HD;       // part*HEADS+head row block

    float s = 0.0f;
#pragma unroll
    for (int d = 0; d < HD; ++d) {
        const float v = Q2[(size_t)(base + d) * NN + n];
        s += v * v;
    }
    const float inv = 1.0f / fmaxf(sqrtf(s), 1e-12f);
#pragma unroll
    for (int d = 0; d < HD; ++d)
        Q2[(size_t)(base + d) * NN + n] *= inv;
}

// ---------------------------------------------------------------------------
// Neighborhood attention. One wave32 per (pixel, head), lane = channel d.
// Writes output DIRECTLY in packed B-fragment f16 layout for the proj GEMM.
// ---------------------------------------------------------------------------
__global__ void na_attention(const float* __restrict__ Q2,
                             const float* __restrict__ rpb,
                             const float* __restrict__ temp,
                             _Float16*    __restrict__ O16)
{
    const int tid  = blockIdx.x * blockDim.x + threadIdx.x;
    const int wave = tid >> 5;
    const int lane = tid & 31;              // channel d (HD == 32)
    if (wave >= NN * HEADS) return;

    const int h   = wave % HEADS;
    const int pix = wave / HEADS;
    const int b   = pix >> 12;
    const int hw  = pix & (HWSZ - 1);
    const int i   = hw >> 6;
    const int j   = hw & 63;
    const int n   = b * HWSZ + hw;

    const int qrow = (0 * HEADS + h) * HD + lane;
    const int krow = (1 * HEADS + h) * HD + lane;
    const int vrow = (2 * HEADS + h) * HD + lane;
    const float q = Q2[(size_t)qrow * NN + n];

    int si = i - (KS / 2); si = si < 0 ? 0 : (si > HH - KS ? HH - KS : si);
    int sj = j - (KS / 2); sj = sj < 0 ? 0 : (sj > WW - KS ? WW - KS : sj);
    const float t = temp[h];

    float logits[KS * KS];
    float mx = -3.4e38f;
#pragma unroll
    for (int ki = 0; ki < KS; ++ki) {
#pragma unroll
        for (int kj = 0; kj < KS; ++kj) {
            const int ii = si + ki;
            const int jj = sj + kj;
            const int nn = b * HWSZ + ii * WW + jj;
            float p = q * Q2[(size_t)krow * NN + nn];
#pragma unroll
            for (int m = 16; m > 0; m >>= 1) p += __shfl_xor(p, m, 32);
            const int ri = ii - i + (KS - 1);
            const int rj = jj - j + (KS - 1);
            const float lg = (p + rpb[(h * (2 * KS - 1) + ri) * (2 * KS - 1) + rj]) * t;
            logits[ki * KS + kj] = lg;
            mx = fmaxf(mx, lg);
        }
    }
    float sum = 0.0f;
#pragma unroll
    for (int e = 0; e < KS * KS; ++e) {
        const float ex = __expf(logits[e] - mx);
        logits[e] = ex;
        sum += ex;
    }
    const float inv = 1.0f / sum;

    float acc = 0.0f;
#pragma unroll
    for (int ki = 0; ki < KS; ++ki) {
#pragma unroll
        for (int kj = 0; kj < KS; ++kj) {
            const int nn = b * HWSZ + (si + ki) * WW + (sj + kj);
            acc += logits[ki * KS + kj] * inv * Q2[(size_t)vrow * NN + nn];
        }
    }

    // write into packed B-fragment layout: row c = h*HD+lane, column n
    const int cch = h * HD + lane;
    const int kt  = cch >> 5;
    const int e   = cch & 15;
    const int lp  = (((cch >> 4) & 1) << 4) | (n & 15);
    const int nt  = n >> 4;
    O16[(((size_t)kt * NT_ALL + nt) * 32 + lp) * 16 + e] = (_Float16)acc;
}

// ---------------------------------------------------------------------------
// Host-side orchestration (graph-capture safe: only stream launches)
// ---------------------------------------------------------------------------
extern "C" void kernel_launch(void* const* d_in, const int* in_sizes, int n_in,
                              void* d_out, int out_size, void* d_ws, size_t ws_size,
                              hipStream_t stream) {
    (void)in_sizes; (void)n_in; (void)out_size; (void)ws_size;

    const float* x      = (const float*)d_in[0];   // (B,C,H,W)
    const float* Wq1    = (const float*)d_in[1];   // (576,192,1,1)
    const float* bq1    = (const float*)d_in[2];   // (576,)
    const float* Wq2    = (const float*)d_in[3];   // (576,1,3,3)
    const float* bq2    = (const float*)d_in[4];   // (576,)
    const float* temp   = (const float*)d_in[5];   // (6,1,1,1)
    const float* rpb    = (const float*)d_in[6];   // (6,13,13)
    const float* Wproj  = (const float*)d_in[7];   // (192,192,1,1)
    const float* bproj  = (const float*)d_in[8];   // (192,)
    float* out = (float*)d_out;

    // ---- workspace carve-up (all offsets 256B-aligned) ----
    char* ws = (char*)d_ws;
    size_t off = 0;
    _Float16* X16  = (_Float16*)(ws + off); off += (size_t)CC * NN * 2;   // 3.0 MB (B-pack)
    _Float16* W1h  = (_Float16*)(ws + off); off += (size_t)C3 * CC * 2;   // 216 KB (A-pack)
    _Float16* WPh  = (_Float16*)(ws + off); off += (size_t)CC * CC * 2;   // 72 KB  (A-pack)
    float*    QKV1 = (float*)   (ws + off); off += (size_t)C3 * NN * 4;   // 18 MB
    float*    QKV2 = (float*)   (ws + off); off += (size_t)C3 * NN * 4;   // 18 MB
    _Float16* O16  = (_Float16*)(ws + off); off += (size_t)CC * NN * 2;   // 3 MB   (B-pack)

    const int TB = 256;

    // 1) pack operands into WMMA fragment layouts (f16)
    na_pack_xB<<<(CC * NN + TB - 1) / TB, TB, 0, stream>>>(x, X16);
    na_pack_wA<<<(C3 * CC + TB - 1) / TB, TB, 0, stream>>>(Wq1, W1h, C3, CC);
    na_pack_wA<<<(CC * CC + TB - 1) / TB, TB, 0, stream>>>(Wproj, WPh, CC, CC);

    // 2) qkv1 1x1 conv as WMMA GEMM: [576x192] x [192x8192] -> QKV1[576][8192]
    {
        const int waves = (C3 / 16) * (NN / 64);   // 16x64 macro-tiles
        na_gemm_wmma<<<(waves * 32 + TB - 1) / TB, TB, 0, stream>>>(
            W1h, X16, bq1, QKV1, C3, NN, CC, /*mode=*/0);
    }

    // 3) depthwise 3x3 + bias
    na_dwconv3x3<<<(C3 * NN + TB - 1) / TB, TB, 0, stream>>>(QKV1, Wq2, bq2, QKV2);

    // 4) l2-normalize q and k rows in place
    na_l2norm<<<(2 * HEADS * NN + TB - 1) / TB, TB, 0, stream>>>(QKV2);

    // 5) neighborhood attention -> O16 already in proj-GEMM B-fragment layout
    na_attention<<<(NN * HEADS * 32 + TB - 1) / TB, TB, 0, stream>>>(QKV2, rpb, temp, O16);

    // 6) proj 1x1 conv as WMMA GEMM, writing final (B,C,H,W)
    {
        const int waves = (CC / 16) * (NN / 64);
        na_gemm_wmma<<<(waves * 32 + TB - 1) / TB, TB, 0, stream>>>(
            WPh, O16, bproj, out, CC, NN, CC, /*mode=*/1);
    }
}
